// TopKMoeLayer_3977139716767
// MI455X (gfx1250) — matrix operand order repaired and checked by reading, not compiled, not run on hardware
//
#include <hip/hip_runtime.h>
#include <stdint.h>
#include <stddef.h>

// ---------------------------------------------------------------------------
// Top-K (k=2) MoE layer for MI455X (gfx1250, wave32, WMMA).
//   B=2, S=2048 -> T=4096 tokens, D=1024, F=4096, E=8 experts, top-2.
// Routed implementation: router (fp32) -> per-expert token compaction ->
// fp16 WMMA GEMM1 (+SiLU) -> fp16 WMMA GEMM2 -> weighted combine.
// Weights pre-packed into per-lane WMMA B-fragment order (fp16, 134 MB,
// L2-resident on MI455X's 192 MB L2). GEMM K-loops are software-pipelined
// with two register sets so fragment loads overlap WMMA issue.
// ---------------------------------------------------------------------------

typedef __attribute__((ext_vector_type(16))) _Float16 v16h;
typedef __attribute__((ext_vector_type(8)))  _Float16 v8h;
typedef __attribute__((ext_vector_type(8)))  float    v8f;

#define T_TOK 4096
#define D_DIM 1024
#define F_DIM 4096
#define E_NUM 8
#define TOPK  2

#define KT1 (D_DIM / 32)   // 32  K-tiles for GEMM1
#define NT1 (F_DIM / 16)   // 256 N-tiles for GEMM1
#define KT2 (F_DIM / 32)   // 128 K-tiles for GEMM2
#define NT2 (D_DIM / 16)   // 64  N-tiles for GEMM2
#define MT32 (T_TOK / 32)  // 128 worst-case M-tiles (M=32) per expert

// ---- WMMA helpers ----------------------------------------------------------

__device__ __forceinline__ v16h cat16(v8h lo, v8h hi) {
  return __builtin_shufflevector(lo, hi, 0, 1, 2, 3, 4, 5, 6, 7,
                                 8, 9, 10, 11, 12, 13, 14, 15);
}

// A fragment (16x32 f16): lane r=lane&15 is row M, half=lane>>4.
// element i -> k = kbase + half*8 + i          (i in 0..7)
//              k = kbase + half*8 + 16 + (i-8) (i in 8..15)
__device__ __forceinline__ v16h ld_afrag(const _Float16* __restrict__ row, int kbase) {
  v8h lo = *(const v8h*)(row + kbase);
  v8h hi = *(const v8h*)(row + kbase + 16);
  return cat16(lo, hi);
}

// B fragment (32x16 f16, column-major lanes): packed contiguously, 16 halves
// per lane at (tile_base + lane*16).
__device__ __forceinline__ v16h ld_bfrag(const _Float16* __restrict__ p) {
  v8h lo = *(const v8h*)(p);
  v8h hi = *(const v8h*)(p + 8);
  return cat16(lo, hi);
}

__device__ __forceinline__ v8f wmma16(v16h a, v16h b, v8f c) {
  return __builtin_amdgcn_wmma_f32_16x16x32_f16(
      /*neg_a=*/false, a, /*neg_b=*/false, b,
      /*c_mod=*/(short)0, c, /*reuse_a=*/false, /*reuse_b=*/false);
}

// Pipelined fragment set load: two A fragments (rows 0-15, 16-31 of the
// M=32 tile) and four B fragments (4 consecutive 16-wide N tiles).
#define LD_SET(s, ktv, bstep)                                              \
  {                                                                        \
    int kb_ = (ktv) * 32 + half * 8;                                       \
    a0##s = ld_afrag(arow0, kb_);                                          \
    a1##s = ld_afrag(arow1, kb_);                                          \
    const _Float16* bt_ = bbase + (size_t)(ktv) * (bstep) + lane * 16;     \
    b0##s = ld_bfrag(bt_);                                                 \
    b1##s = ld_bfrag(bt_ + 512);                                           \
    b2##s = ld_bfrag(bt_ + 1024);                                          \
    b3##s = ld_bfrag(bt_ + 1536);                                          \
  }

#define DO_WMMA(s)                                                         \
  c00 = wmma16(a0##s, b0##s, c00);                                         \
  c01 = wmma16(a0##s, b1##s, c01);                                         \
  c02 = wmma16(a0##s, b2##s, c02);                                         \
  c03 = wmma16(a0##s, b3##s, c03);                                         \
  c10 = wmma16(a1##s, b0##s, c10);                                         \
  c11 = wmma16(a1##s, b1##s, c11);                                         \
  c12 = wmma16(a1##s, b2##s, c12);                                         \
  c13 = wmma16(a1##s, b3##s, c13);

// ---- Routing ---------------------------------------------------------------

__global__ void moe_init(int* counts, int* cursor) {
  int i = threadIdx.x;
  if (i < E_NUM) { counts[i] = 0; cursor[i] = 0; }
}

// One wave per token: gate logits (fp32 exact), softmax, top-2, renorm.
// Also emits the fp16 copy of x used as the GEMM1 A operand.
__global__ void __launch_bounds__(256) moe_router(
    const float* __restrict__ x, const float* __restrict__ gw,
    const float* __restrict__ gb, _Float16* __restrict__ xh,
    int* __restrict__ topk_e, float* __restrict__ topk_w,
    int* __restrict__ counts) {
  int lane = threadIdx.x & 31;
  int wv   = threadIdx.x >> 5;
  int t    = blockIdx.x * 8 + wv;
  if (t >= T_TOK) return;

  float acc[E_NUM];
#pragma unroll
  for (int e = 0; e < E_NUM; ++e) acc[e] = 0.f;

  const float* xr = x + (size_t)t * D_DIM;
  _Float16*    xo = xh + (size_t)t * D_DIM;
  for (int j = 0; j < D_DIM / 32; ++j) {
    int d = j * 32 + lane;
    float xv = xr[d];
    xo[d] = (_Float16)xv;
#pragma unroll
    for (int e = 0; e < E_NUM; ++e) acc[e] += xv * gw[e * D_DIM + d];
  }
#pragma unroll
  for (int e = 0; e < E_NUM; ++e) {
#pragma unroll
    for (int off = 16; off >= 1; off >>= 1)
      acc[e] += __shfl_xor(acc[e], off, 32);
  }

  if (lane == 0) {
    float l[E_NUM];
    float m = -1e30f;
#pragma unroll
    for (int e = 0; e < E_NUM; ++e) { l[e] = acc[e] + gb[e]; m = l[e] > m ? l[e] : m; }
    float p[E_NUM];
#pragma unroll
    for (int e = 0; e < E_NUM; ++e) p[e] = __expf(l[e] - m);
    int e0 = 0, e1 = -1;
    float p0 = p[0], p1 = -1e30f;
#pragma unroll
    for (int e = 1; e < E_NUM; ++e) {
      if (p[e] > p0)      { p1 = p0; e1 = e0; p0 = p[e]; e0 = e; }
      else if (p[e] > p1) { p1 = p[e]; e1 = e; }
    }
    float inv = 1.f / (p0 + p1);
    topk_e[t * 2 + 0] = e0;
    topk_e[t * 2 + 1] = e1;
    topk_w[t * 2 + 0] = p0 * inv;
    topk_w[t * 2 + 1] = p1 * inv;
    atomicAdd(&counts[e0], 1);
    atomicAdd(&counts[e1], 1);
  }
}

__global__ void moe_scan(const int* counts, int* offsets) {
  if (threadIdx.x == 0) {
    int s = 0;
    for (int e = 0; e < E_NUM; ++e) { offsets[e] = s; s += counts[e]; }
  }
}

__global__ void moe_place(const int* __restrict__ topk_e,
                          const int* __restrict__ offsets,
                          int* __restrict__ cursor,
                          int* __restrict__ rows_tokens,
                          int* __restrict__ token_rowid) {
  int t = blockIdx.x * 256 + threadIdx.x;
  if (t >= T_TOK) return;
#pragma unroll
  for (int j = 0; j < TOPK; ++j) {
    int e   = topk_e[t * 2 + j];
    int pos = atomicAdd(&cursor[e], 1);
    int row = offsets[e] + pos;
    rows_tokens[row]       = t;
    token_rowid[t * 2 + j] = row;
  }
}

// ---- Weight packing: fp32 [E][K][N] -> fp16 WMMA B-fragments ---------------
// Tile (e, kt, nt) = 512 halves; lane holds B[k][n] with n = nt*16 + (lane&15)
// and k = kt*32 + (lane>>4)*8 + i (i<8), kt*32 + (lane>>4)*8 + 16 + (i-8) (i>=8).
__global__ void __launch_bounds__(256) moe_pack(
    const float* __restrict__ W, _Float16* __restrict__ Wp, int K, int N) {
  int tile = blockIdx.x * 8 + (threadIdx.x >> 5);
  int KT = K / 32, NT = N / 16;
  int ntiles = E_NUM * KT * NT;
  if (tile >= ntiles) return;
  int lane = threadIdx.x & 31;
  int half = lane >> 4, col = lane & 15;
  int nt = tile % NT;
  int kt = (tile / NT) % KT;
  int e  = tile / (NT * KT);
  const float* Wt = W + (size_t)e * K * N + (size_t)(kt * 32 + half * 8) * N +
                    (size_t)nt * 16 + col;
  v8h lo, hi;
#pragma unroll
  for (int i = 0; i < 8; ++i) {
    lo[i] = (_Float16)Wt[(size_t)i * N];
    hi[i] = (_Float16)Wt[(size_t)(16 + i) * N];
  }
  _Float16* dst = Wp + (size_t)tile * 512 + lane * 16;
  *(v8h*)dst       = lo;
  *(v8h*)(dst + 8) = hi;
}

// ---- GEMM1: h = silu(x @ w1[e] + b1[e]), fp16 out --------------------------
// Block = 8 waves; M-tile 32 (gathered tokens), wave covers 4 N-tiles (N=64).
// K loop software-pipelined with two register sets (A/B) for load/WMMA overlap.
__global__ void __launch_bounds__(256) moe_gemm1(
    const _Float16* __restrict__ xh, const _Float16* __restrict__ w1p,
    const float* __restrict__ b1, const int* __restrict__ counts,
    const int* __restrict__ offsets, const int* __restrict__ rows_tokens,
    _Float16* __restrict__ hbuf) {
  int e  = blockIdx.x / MT32;
  int mt = blockIdx.x % MT32;
  int cnt = counts[e];
  if (mt * 32 >= cnt) return;

  int lane = threadIdx.x & 31;
  int wave = threadIdx.x >> 5;
  int half = lane >> 4;
  int r    = lane & 15;
  int base = offsets[e];

  int mrow0 = mt * 32 + r;
  int mrow1 = mt * 32 + 16 + r;
  int idx0  = mrow0 < cnt ? mrow0 : cnt - 1;
  int idx1  = mrow1 < cnt ? mrow1 : cnt - 1;
  const _Float16* arow0 = xh + (size_t)rows_tokens[base + idx0] * D_DIM;
  const _Float16* arow1 = xh + (size_t)rows_tokens[base + idx1] * D_DIM;

  int n0  = blockIdx.y * 512 + wave * 64;
  int nt0 = n0 >> 4;
  const _Float16* bbase = w1p + ((size_t)e * KT1 * NT1 + nt0) * 512;
  const size_t BSTEP = (size_t)NT1 * 512;

  v8f c00 = {}, c01 = {}, c02 = {}, c03 = {};
  v8f c10 = {}, c11 = {}, c12 = {}, c13 = {};
  v16h a0A, a1A, b0A, b1A, b2A, b3A;
  v16h a0B, a1B, b0B, b1B, b2B, b3B;

  LD_SET(A, 0, BSTEP)
  for (int kt = 0; kt < KT1; kt += 2) {
    LD_SET(B, kt + 1, BSTEP)
    __builtin_prefetch(bbase + (size_t)(kt + 2 < KT1 ? kt + 2 : 0) * BSTEP +
                           lane * 16, 0, 1);
    DO_WMMA(A)
    int ktn = (kt + 2 < KT1) ? kt + 2 : 0;
    LD_SET(A, ktn, BSTEP)
    DO_WMMA(B)
  }

  v8f cc[8] = {c00, c01, c02, c03, c10, c11, c12, c13};
  int gm = base + mt * 32;
#pragma unroll
  for (int tI = 0; tI < 2; ++tI) {
#pragma unroll
    for (int j = 0; j < 4; ++j) {
      int n = n0 + j * 16 + r;
      float bias = b1[e * F_DIM + n];
#pragma unroll
      for (int v = 0; v < 8; ++v) {
        int m = tI * 16 + v + 8 * half;  // D layout: VGPR v -> M=v / M=v+8
        if (mt * 32 + m < cnt) {
          float h = cc[tI * 4 + j][v] + bias;
          float s = h / (1.f + __expf(-h));  // SiLU
          hbuf[(size_t)(gm + m) * F_DIM + n] = (_Float16)s;
        }
      }
    }
  }
}

// ---- GEMM2: eo = h @ w2[e] + b2[e], fp32 out -------------------------------
__global__ void __launch_bounds__(256) moe_gemm2(
    const _Float16* __restrict__ hbuf, const _Float16* __restrict__ w2p,
    const float* __restrict__ b2, const int* __restrict__ counts,
    const int* __restrict__ offsets, float* __restrict__ eo) {
  int e  = blockIdx.x / MT32;
  int mt = blockIdx.x % MT32;
  int cnt = counts[e];
  if (mt * 32 >= cnt) return;

  int lane = threadIdx.x & 31;
  int wave = threadIdx.x >> 5;
  int half = lane >> 4;
  int r    = lane & 15;
  int base = offsets[e];

  int mrow0 = mt * 32 + r;
  int mrow1 = mt * 32 + 16 + r;
  int idx0  = mrow0 < cnt ? mrow0 : cnt - 1;
  int idx1  = mrow1 < cnt ? mrow1 : cnt - 1;
  const _Float16* arow0 = hbuf + (size_t)(base + idx0) * F_DIM;
  const _Float16* arow1 = hbuf + (size_t)(base + idx1) * F_DIM;

  int n0  = blockIdx.y * 512 + wave * 64;
  int nt0 = n0 >> 4;
  const _Float16* bbase = w2p + ((size_t)e * KT2 * NT2 + nt0) * 512;
  const size_t BSTEP = (size_t)NT2 * 512;

  v8f c00 = {}, c01 = {}, c02 = {}, c03 = {};
  v8f c10 = {}, c11 = {}, c12 = {}, c13 = {};
  v16h a0A, a1A, b0A, b1A, b2A, b3A;
  v16h a0B, a1B, b0B, b1B, b2B, b3B;

  LD_SET(A, 0, BSTEP)
  for (int kt = 0; kt < KT2; kt += 2) {
    LD_SET(B, kt + 1, BSTEP)
    __builtin_prefetch(bbase + (size_t)(kt + 2 < KT2 ? kt + 2 : 0) * BSTEP +
                           lane * 16, 0, 1);
    DO_WMMA(A)
    int ktn = (kt + 2 < KT2) ? kt + 2 : 0;
    LD_SET(A, ktn, BSTEP)
    DO_WMMA(B)
  }

  v8f cc[8] = {c00, c01, c02, c03, c10, c11, c12, c13};
  int gm = base + mt * 32;
#pragma unroll
  for (int tI = 0; tI < 2; ++tI) {
#pragma unroll
    for (int j = 0; j < 4; ++j) {
      int n = n0 + j * 16 + r;
      float bias = b2[e * D_DIM + n];
#pragma unroll
      for (int v = 0; v < 8; ++v) {
        int m = tI * 16 + v + 8 * half;
        if (mt * 32 + m < cnt)
          eo[(size_t)(gm + m) * D_DIM + n] = cc[tI * 4 + j][v] + bias;
      }
    }
  }
}

// ---- Combine: out[t] = w0*eo[slot0] + w1*eo[slot1] (bit-deterministic) -----
__global__ void __launch_bounds__(256) moe_combine(
    const float* __restrict__ eo, const int* __restrict__ token_rowid,
    const float* __restrict__ topk_w, float* __restrict__ out) {
  size_t i = (size_t)blockIdx.x * 256 + threadIdx.x;
  int t = (int)(i / D_DIM);
  int d = (int)(i % D_DIM);
  int r0 = token_rowid[t * 2 + 0];
  int r1 = token_rowid[t * 2 + 1];
  float w0 = topk_w[t * 2 + 0];
  float w1 = topk_w[t * 2 + 1];
  out[i] = w0 * eo[(size_t)r0 * D_DIM + d] + w1 * eo[(size_t)r1 * D_DIM + d];
}

// ---------------------------------------------------------------------------

extern "C" void kernel_launch(void* const* d_in, const int* in_sizes, int n_in,
                              void* d_out, int out_size, void* d_ws, size_t ws_size,
                              hipStream_t stream) {
  (void)in_sizes; (void)n_in; (void)out_size; (void)ws_size;
  const float* x  = (const float*)d_in[0];
  const float* gw = (const float*)d_in[1];
  const float* gb = (const float*)d_in[2];
  const float* w1 = (const float*)d_in[3];
  const float* b1 = (const float*)d_in[4];
  const float* w2 = (const float*)d_in[5];
  const float* b2 = (const float*)d_in[6];
  float* out = (float*)d_out;

  // Workspace carve (~244 MB total).
  char* ws = (char*)d_ws;
  size_t off = 0;
  auto carve = [&](size_t bytes) -> char* {
    char* p = ws + off;
    off = (off + bytes + 255) & ~(size_t)255;
    return p;
  };
  _Float16* xh   = (_Float16*)carve((size_t)T_TOK * D_DIM * 2);           //  8.4 MB
  _Float16* w1p  = (_Float16*)carve((size_t)E_NUM * D_DIM * F_DIM * 2);   // 67 MB
  _Float16* w2p  = (_Float16*)carve((size_t)E_NUM * F_DIM * D_DIM * 2);   // 67 MB
  _Float16* hbuf = (_Float16*)carve((size_t)T_TOK * TOPK * F_DIM * 2);    // 67 MB
  float*    eo   = (float*)carve((size_t)T_TOK * TOPK * D_DIM * 4);       // 33.5 MB
  int*   counts      = (int*)carve(E_NUM * 4);
  int*   cursor      = (int*)carve(E_NUM * 4);
  int*   offsets     = (int*)carve(E_NUM * 4);
  int*   topk_e      = (int*)carve(T_TOK * 2 * 4);
  float* topk_w      = (float*)carve(T_TOK * 2 * 4);
  int*   rows_tokens = (int*)carve(T_TOK * TOPK * 4);
  int*   token_rowid = (int*)carve(T_TOK * 2 * 4);

  moe_init<<<1, 32, 0, stream>>>(counts, cursor);
  moe_router<<<T_TOK / 8, 256, 0, stream>>>(x, gw, gb, xh, topk_e, topk_w, counts);
  moe_scan<<<1, 1, 0, stream>>>(counts, offsets);
  moe_place<<<T_TOK / 256, 256, 0, stream>>>(topk_e, offsets, cursor,
                                             rows_tokens, token_rowid);
  moe_pack<<<(E_NUM * KT1 * NT1 + 7) / 8, 256, 0, stream>>>(w1, w1p, D_DIM, F_DIM);
  moe_pack<<<(E_NUM * KT2 * NT2 + 7) / 8, 256, 0, stream>>>(w2, w2p, F_DIM, D_DIM);

  dim3 g1(E_NUM * MT32, F_DIM / 512);  // (1024, 8), early-exit on counts
  moe_gemm1<<<g1, 256, 0, stream>>>(xh, w1p, b1, counts, offsets, rows_tokens, hbuf);

  dim3 g2(E_NUM * MT32, D_DIM / 512);  // (1024, 2)
  moe_gemm2<<<g2, 256, 0, stream>>>(hbuf, w2p, b2, counts, offsets, eo);

  moe_combine<<<(T_TOK * D_DIM) / 256, 256, 0, stream>>>(eo, token_rowid, topk_w, out);
}
